// EGC_30940944401178
// MI455X (gfx1250) — matrix operand order repaired and checked by reading, not compiled, not run on hardware
//
#include <hip/hip_runtime.h>
#include <hip/hip_bf16.h>

typedef __attribute__((ext_vector_type(16))) _Float16 v16h;
typedef __attribute__((ext_vector_type(8)))  float    v8f;

#define N_NODES 100000
#define N_EDGES 1600000
#define ETILES  (N_EDGES / 16)   // 100000
#define NTILES  (N_NODES / 16)   // 6250
#define AW_DWORDS (16 * 256)     // 16 A-tiles, 256 dwords each
#define AW_BYTES  (AW_DWORDS * 4)

// ---------------------------------------------------------------------------
// Weight prep: pack W^T tiles into the 16-bit WMMA A-operand layout.
// A-layout (contiguous k per lane half, 16-bit A 16x32): lane = 16*half + m;
//   lanes 0-15 : V0-7 -> K 0..15 (pairs)
//   lanes 16-31: V0-7 -> K 16..31
// Tile table (all W are [K][32] row-major):
//   0-5 : ew1 (K=72)  (mt,kt) = (0,0)(0,1)(0,2)(1,0)(1,1)(1,2)
//   6-7 : ew2 (K=32)  mt 0,1
//   8-9 : cw1 (K=32)  mt 0,1
//   10-13: nw1 (K=64) (0,0)(0,1)(1,0)(1,1)
//   14-15: nw2 (K=32) mt 0,1
// ---------------------------------------------------------------------------
__global__ __launch_bounds__(256) void egc_prep_weights(
    const float* __restrict__ ew1, const float* __restrict__ ew2,
    const float* __restrict__ cw1, const float* __restrict__ nw1,
    const float* __restrict__ nw2, unsigned int* __restrict__ aw) {
  int tid = blockIdx.x * blockDim.x + threadIdx.x;
  if (tid >= AW_DWORDS) return;
  int tile = tid >> 8;
  int rem  = tid & 255;
  int lane = rem >> 3;
  int v    = rem & 7;
  int m    = lane & 15;
  int hh   = lane >> 4;

  const float* W; int mt, kt, Kdim;
  switch (tile) {
    case 0:  W = ew1; mt = 0; kt = 0; Kdim = 72; break;
    case 1:  W = ew1; mt = 0; kt = 1; Kdim = 72; break;
    case 2:  W = ew1; mt = 0; kt = 2; Kdim = 72; break;
    case 3:  W = ew1; mt = 1; kt = 0; Kdim = 72; break;
    case 4:  W = ew1; mt = 1; kt = 1; Kdim = 72; break;
    case 5:  W = ew1; mt = 1; kt = 2; Kdim = 72; break;
    case 6:  W = ew2; mt = 0; kt = 0; Kdim = 32; break;
    case 7:  W = ew2; mt = 1; kt = 0; Kdim = 32; break;
    case 8:  W = cw1; mt = 0; kt = 0; Kdim = 32; break;
    case 9:  W = cw1; mt = 1; kt = 0; Kdim = 32; break;
    case 10: W = nw1; mt = 0; kt = 0; Kdim = 64; break;
    case 11: W = nw1; mt = 0; kt = 1; Kdim = 64; break;
    case 12: W = nw1; mt = 1; kt = 0; Kdim = 64; break;
    case 13: W = nw1; mt = 1; kt = 1; Kdim = 64; break;
    case 14: W = nw2; mt = 0; kt = 0; Kdim = 32; break;
    default: W = nw2; mt = 1; kt = 0; Kdim = 32; break;
  }

  unsigned int out = 0;
  #pragma unroll
  for (int s = 0; s < 2; ++s) {
    int k  = hh * 16 + v * 2 + s;
    int kg = kt * 32 + k;
    int mg = mt * 16 + m;
    float val = (kg < Kdim) ? W[kg * 32 + mg] : 0.0f;
    union { _Float16 f; unsigned short b; } cv; cv.f = (_Float16)val;
    out |= ((unsigned int)cv.b) << (16 * s);
  }
  aw[tile * 256 + lane * 8 + v] = out;
}

// ---------------------------------------------------------------------------
// helpers
// ---------------------------------------------------------------------------
__device__ inline v16h egc_load_aw(const unsigned int* __restrict__ aw, int tile, int lane) {
  union { uint4 q[2]; v16h h; } u;
  const uint4* p = (const uint4*)(aw + tile * 256 + lane * 8);
  u.q[0] = p[0]; u.q[1] = p[1];
  return u.h;
}

__device__ inline v8f egc_load_bias8(const float* __restrict__ b) {
  union { float4 q[2]; v8f f; } u;
  const float4* p = (const float4*)b;
  u.q[0] = p[0]; u.q[1] = p[1];
  return u.f;
}

// 16 consecutive f32 -> f16 B-operand column (B layout: lane holds its
// K-half contiguously: k_local = 2v+s, element idx == k_local)
__device__ inline v16h egc_load_row16(const float* __restrict__ p) {
  const float4* q = (const float4*)p;
  float4 a = q[0], b = q[1], c = q[2], d = q[3];
  v16h r;
  r[0]=(_Float16)a.x; r[1]=(_Float16)a.y; r[2]=(_Float16)a.z; r[3]=(_Float16)a.w;
  r[4]=(_Float16)b.x; r[5]=(_Float16)b.y; r[6]=(_Float16)b.z; r[7]=(_Float16)b.w;
  r[8]=(_Float16)c.x; r[9]=(_Float16)c.y; r[10]=(_Float16)c.z; r[11]=(_Float16)c.w;
  r[12]=(_Float16)d.x; r[13]=(_Float16)d.y; r[14]=(_Float16)d.z; r[15]=(_Float16)d.w;
  return r;
}

__device__ inline float egc_silu(float x) { return x / (1.0f + __expf(-x)); }

// Convert two 16x16 D tiles (feat-major rows, edge columns) into the next
// layer's B operand. D layout: lane L holds column n=L%16, rows m = v+8*(L/16).
// B lane L needs k = 16*(L/16) + 0..15 of edge L%16 -> half-swap via shfl_xor 16.
__device__ inline v16h egc_build_b(const float d0[8], const float d1[8], int h) {
  v16h b;
  #pragma unroll
  for (int v = 0; v < 8; ++v) {
    float x0 = d0[v], x1 = d1[v];
    float p0 = __shfl_xor(x0, 16, 32);
    float p1 = __shfl_xor(x1, 16, 32);
    float lo = h ? p1 : x0;   // k_local = v
    float hi = h ? x1 : p0;   // k_local = 8+v
    b[v]     = (_Float16)lo;
    b[8 + v] = (_Float16)hi;
  }
  return b;
}

#define WMMA(a, b, c) __builtin_amdgcn_wmma_f32_16x16x32_f16( \
    false, (a), false, (b), (short)0, (c), false, false)

// ---------------------------------------------------------------------------
// Edge kernel: one wave = 16 edges. D[16 outfeat][16 edges] = W^T x act.
// ---------------------------------------------------------------------------
__global__ __launch_bounds__(256) void egc_edge_kernel(
    const float* __restrict__ coord, const float* __restrict__ nf,
    const int* __restrict__ eidx, const float* __restrict__ ffc,
    const float* __restrict__ eb1, const float* __restrict__ eb2,
    const float* __restrict__ cb1, const float* __restrict__ cw2,
    const unsigned int* __restrict__ aw,
    float* __restrict__ agg_h, float* __restrict__ agg_c,
    float* __restrict__ cnt) {
  const int lane  = threadIdx.x & 31;
  const int e_loc = lane & 15;
  const int h     = lane >> 4;
  const int wid     = blockIdx.x * (blockDim.x >> 5) + (threadIdx.x >> 5);
  const int wstride = gridDim.x * (blockDim.x >> 5);

  // resident weight A-operands
  v16h a_e1[2][3], a_e2[2], a_c1[2];
  #pragma unroll
  for (int t = 0; t < 2; ++t)
    #pragma unroll
    for (int k = 0; k < 3; ++k) a_e1[t][k] = egc_load_aw(aw, t * 3 + k, lane);
  a_e2[0] = egc_load_aw(aw, 6, lane); a_e2[1] = egc_load_aw(aw, 7, lane);
  a_c1[0] = egc_load_aw(aw, 8, lane); a_c1[1] = egc_load_aw(aw, 9, lane);

  // cw2 slices matching the D layout this lane will hold
  float cwa[8], cwb[8];
  #pragma unroll
  for (int v = 0; v < 8; ++v) {
    cwa[v] = cw2[8 * h + v];
    cwb[v] = cw2[16 + 8 * h + v];
  }
  const float fc0 = ffc[0], fc1 = ffc[1], fc2 = ffc[2], fc3 = ffc[3];
  const float TWO_PI = 6.283185307179586f;

  for (int tile = wid; tile < ETILES; tile += wstride) {
    const int e  = tile * 16 + e_loc;
    const int ii = eidx[e];
    const int jj = eidx[N_EDGES + e];

    const float dx = coord[ii * 3 + 0] - coord[jj * 3 + 0];
    const float dy = coord[ii * 3 + 1] - coord[jj * 3 + 1];
    const float dz = coord[ii * 3 + 2] - coord[jj * 3 + 2];
    const float radial = dx * dx + dy * dy + dz * dz;
    const float xr = TWO_PI * radial;
    float ff[8];
    ff[0] = __cosf(xr * fc0); ff[1] = __cosf(xr * fc1);
    ff[2] = __cosf(xr * fc2); ff[3] = __cosf(xr * fc3);
    ff[4] = __sinf(xr * fc0); ff[5] = __sinf(xr * fc1);
    ff[6] = __sinf(xr * fc2); ff[7] = __sinf(xr * fc3);

    // layer-1 B operands: gathered node features + fourier tile
    v16h b0 = egc_load_row16(nf + ii * 32 + h * 16);
    v16h b1 = egc_load_row16(nf + jj * 32 + h * 16);
    v16h b2;
    #pragma unroll
    for (int t = 0; t < 8; ++t) {
      b2[t]     = (h == 0) ? (_Float16)ff[t] : (_Float16)0.0f;
      b2[8 + t] = (_Float16)0.0f;
    }

    // edge MLP layer 1 (K = 96 padded)
    float m1a[8], m1b[8];
    {
      v8f c = egc_load_bias8(eb1 + 8 * h);
      c = WMMA(a_e1[0][0], b0, c);
      c = WMMA(a_e1[0][1], b1, c);
      c = WMMA(a_e1[0][2], b2, c);
      #pragma unroll
      for (int v = 0; v < 8; ++v) m1a[v] = egc_silu(c[v]);
    }
    {
      v8f c = egc_load_bias8(eb1 + 16 + 8 * h);
      c = WMMA(a_e1[1][0], b0, c);
      c = WMMA(a_e1[1][1], b1, c);
      c = WMMA(a_e1[1][2], b2, c);
      #pragma unroll
      for (int v = 0; v < 8; ++v) m1b[v] = egc_silu(c[v]);
    }

    // edge MLP layer 2
    v16h bm1 = egc_build_b(m1a, m1b, h);
    float m2a[8], m2b[8];
    {
      v8f c = egc_load_bias8(eb2 + 8 * h);
      c = WMMA(a_e2[0], bm1, c);
      #pragma unroll
      for (int v = 0; v < 8; ++v) m2a[v] = egc_silu(c[v]);
    }
    {
      v8f c = egc_load_bias8(eb2 + 16 + 8 * h);
      c = WMMA(a_e2[1], bm1, c);
      #pragma unroll
      for (int v = 0; v < 8; ++v) m2b[v] = egc_silu(c[v]);
    }

    // coord head layer 1
    v16h bm2 = egc_build_b(m2a, m2b, h);
    float t1a[8], t1b[8];
    {
      v8f c = egc_load_bias8(cb1 + 8 * h);
      c = WMMA(a_c1[0], bm2, c);
      #pragma unroll
      for (int v = 0; v < 8; ++v) t1a[v] = egc_silu(c[v]);
    }
    {
      v8f c = egc_load_bias8(cb1 + 16 + 8 * h);
      c = WMMA(a_c1[1], bm2, c);
      #pragma unroll
      for (int v = 0; v < 8; ++v) t1b[v] = egc_silu(c[v]);
    }

    // w[e] = t1 . cw2  (partial per lane-half, combine via xor-16)
    float wp = 0.0f;
    #pragma unroll
    for (int v = 0; v < 8; ++v) wp += t1a[v] * cwa[v] + t1b[v] * cwb[v];
    const float w = wp + __shfl_xor(wp, 16, 32);

    // aggregate messages (sum) into nodes
    const int hbase = ii * 32;
    #pragma unroll
    for (int v = 0; v < 8; ++v)
      atomicAdd(&agg_h[hbase + 8 * h + v], m2a[v]);
    #pragma unroll
    for (int v = 0; v < 8; ++v)
      atomicAdd(&agg_h[hbase + 16 + 8 * h + v], m2b[v]);

    if (h == 0) {
      atomicAdd(&agg_c[ii * 3 + 0], dx * w);
      atomicAdd(&agg_c[ii * 3 + 1], dy * w);
      atomicAdd(&agg_c[ii * 3 + 2], dz * w);
      atomicAdd(&cnt[ii], 1.0f);
    }
  }
}

// ---------------------------------------------------------------------------
// Node kernel: [nf | agg_h] -> silu -> nw2 + residual
// ---------------------------------------------------------------------------
__global__ __launch_bounds__(256) void egc_node_kernel(
    const float* __restrict__ nf, const float* __restrict__ agg_h,
    const float* __restrict__ nb1, const float* __restrict__ nb2,
    const unsigned int* __restrict__ aw, float* __restrict__ out) {
  const int lane  = threadIdx.x & 31;
  const int n_loc = lane & 15;
  const int h     = lane >> 4;
  const int wid     = blockIdx.x * (blockDim.x >> 5) + (threadIdx.x >> 5);
  const int wstride = gridDim.x * (blockDim.x >> 5);

  v16h a_n1[2][2], a_n2[2];
  #pragma unroll
  for (int t = 0; t < 2; ++t)
    #pragma unroll
    for (int k = 0; k < 2; ++k) a_n1[t][k] = egc_load_aw(aw, 10 + t * 2 + k, lane);
  a_n2[0] = egc_load_aw(aw, 14, lane);
  a_n2[1] = egc_load_aw(aw, 15, lane);

  for (int tile = wid; tile < NTILES; tile += wstride) {
    const int n = tile * 16 + n_loc;
    v16h b0 = egc_load_row16(nf + n * 32 + h * 16);
    v16h b1 = egc_load_row16(agg_h + n * 32 + h * 16);

    float h1a[8], h1b[8];
    {
      v8f c = egc_load_bias8(nb1 + 8 * h);
      c = WMMA(a_n1[0][0], b0, c);
      c = WMMA(a_n1[0][1], b1, c);
      #pragma unroll
      for (int v = 0; v < 8; ++v) h1a[v] = egc_silu(c[v]);
    }
    {
      v8f c = egc_load_bias8(nb1 + 16 + 8 * h);
      c = WMMA(a_n1[1][0], b0, c);
      c = WMMA(a_n1[1][1], b1, c);
      #pragma unroll
      for (int v = 0; v < 8; ++v) h1b[v] = egc_silu(c[v]);
    }

    v16h bh = egc_build_b(h1a, h1b, h);

    #pragma unroll
    for (int t = 0; t < 2; ++t) {
      v8f c = egc_load_bias8(nb2 + t * 16 + 8 * h);
      c = (t == 0) ? WMMA(a_n2[0], bh, c) : WMMA(a_n2[1], bh, c);
      const int base = n * 32 + t * 16 + 8 * h;
      float4 r0 = *(const float4*)(nf + base);
      float4 r1 = *(const float4*)(nf + base + 4);
      float4 o0, o1;
      o0.x = c[0] + r0.x; o0.y = c[1] + r0.y; o0.z = c[2] + r0.z; o0.w = c[3] + r0.w;
      o1.x = c[4] + r1.x; o1.y = c[5] + r1.y; o1.z = c[6] + r1.z; o1.w = c[7] + r1.w;
      *(float4*)(out + base)     = o0;
      *(float4*)(out + base + 4) = o1;
    }
  }
}

// ---------------------------------------------------------------------------
// Coord finalize: coord + agg_c / max(cnt, 1)
// ---------------------------------------------------------------------------
__global__ __launch_bounds__(256) void egc_coord_kernel(
    const float* __restrict__ coord, const float* __restrict__ agg_c,
    const float* __restrict__ cnt, float* __restrict__ out) {
  int n = blockIdx.x * blockDim.x + threadIdx.x;
  if (n >= N_NODES) return;
  float inv = 1.0f / fmaxf(cnt[n], 1.0f);
  #pragma unroll
  for (int d = 0; d < 3; ++d)
    out[n * 3 + d] = coord[n * 3 + d] + agg_c[n * 3 + d] * inv;
}

// ---------------------------------------------------------------------------
extern "C" void kernel_launch(void* const* d_in, const int* in_sizes, int n_in,
                              void* d_out, int out_size, void* d_ws, size_t ws_size,
                              hipStream_t stream) {
  const float* coord = (const float*)d_in[0];
  const float* nfeat = (const float*)d_in[1];
  const int*   eidx  = (const int*)d_in[2];
  const float* ffc   = (const float*)d_in[3];
  const float* ew1   = (const float*)d_in[4];
  const float* eb1   = (const float*)d_in[5];
  const float* ew2   = (const float*)d_in[6];
  const float* eb2   = (const float*)d_in[7];
  const float* cw1   = (const float*)d_in[8];
  const float* cb1   = (const float*)d_in[9];
  const float* cw2   = (const float*)d_in[10];
  const float* nw1   = (const float*)d_in[11];
  const float* nb1   = (const float*)d_in[12];
  const float* nw2   = (const float*)d_in[13];
  const float* nb2   = (const float*)d_in[14];

  unsigned int* aw = (unsigned int*)d_ws;
  float* agg_h = (float*)((char*)d_ws + AW_BYTES);
  float* agg_c = agg_h + (size_t)N_NODES * 32;
  float* cnt   = agg_c + (size_t)N_NODES * 3;
  float* nout  = (float*)d_out;
  float* cout  = nout + (size_t)N_NODES * 32;

  // zero the aggregation buffers every call (deterministic)
  hipMemsetAsync(agg_h, 0, (size_t)(N_NODES * 32 + N_NODES * 3 + N_NODES) * sizeof(float), stream);

  egc_prep_weights<<<16, 256, 0, stream>>>(ew1, ew2, cw1, nw1, nw2, aw);
  egc_edge_kernel<<<2048, 256, 0, stream>>>(coord, nfeat, eidx, ffc,
                                            eb1, eb2, cb1, cw2, aw,
                                            agg_h, agg_c, cnt);
  egc_node_kernel<<<782, 256, 0, stream>>>(nfeat, agg_h, nb1, nb2, aw, nout);
  egc_coord_kernel<<<(N_NODES + 255) / 256, 256, 0, stream>>>(coord, agg_c, cnt, cout);
}
